// CRF_81286551044351
// MI455X (gfx1250) — matrix operand order repaired and checked by reading, not compiled, not run on hardware
//
#include <hip/hip_runtime.h>

// CRF log-likelihood (SEQ=1024, BATCH=512, NTAGS=64) on gfx1250 / MI455X.
// Normalizer: forward algorithm as per-step 64x64x16 matmul in probability
// space (E = exp(transitions) constant), via v_wmma_f32_16x16x32_f16 with f32
// accumulate. Scores kept in base-2 domain, normalized to max=0 with a running
// per-row offset -> raw v_exp_f32 / v_log_f32 (no denorm guards), and the
// step-opening exp2(z) has no dependency on the current step's memory loads.

#define SEQ   1024
#define BATCH 512
#define NTAGS 64
#define LOG2E 1.4426950408889634f
#define LN2F  0.6931471805599453f

typedef float    v4f  __attribute__((ext_vector_type(4)));
typedef float    v8f  __attribute__((ext_vector_type(8)));
typedef __fp16   h2   __attribute__((ext_vector_type(2)));
typedef __fp16   h8   __attribute__((ext_vector_type(8)));
typedef _Float16 v16h __attribute__((ext_vector_type(16)));

union PackH8 { h2 q[4]; h8 v; };
union Frag16 { h8 part[2]; v16h v; };

// ---------------------------------------------------------------------------
// Forward (normalizer): each wave owns 16 batch rows; 8 waves per block.
// State layout (C/D layout of D = E^T x P^T):
//   lane (h,l16), vgpr v, tile mt -> z[b=b0+l16][j = mt*16 + 8h + v]
// where z = score*log2(e) - o, max_j z = 0, o = per-row offset (per lane).
// ---------------------------------------------------------------------------
__global__ __launch_bounds__(256)
void crf_forward_kernel(const float* __restrict__ em,
                        const int*   __restrict__ mask,
                        const float* __restrict__ startT,
                        const float* __restrict__ endT,
                        const float* __restrict__ trans,
                        float*       __restrict__ den_out) {
  __shared__ __align__(16) __fp16 plds[8][16][72];  // per-wave [b][tag] tile
  const int lane = threadIdx.x & 31;
  const int w    = threadIdx.x >> 5;
  const int h    = lane >> 4;
  const int l16  = lane & 15;
  const int b    = (blockIdx.x * 8 + w) * 16 + l16;

  // ---- A = E^T fragments (constant), f16. A 16x32 layout: lane M=l16,
  // elements 0..7 -> K = 32kc + 8h + e ; elements 8..15 -> K = 32kc+16+8h+(e-8)
  v16h efrag[4][2];
#pragma unroll
  for (int kc = 0; kc < 2; ++kc)
#pragma unroll
    for (int mt = 0; mt < 4; ++mt)
#pragma unroll
      for (int e = 0; e < 16; ++e) {
        const int i = 32 * kc + ((e < 8) ? (8 * h + e) : (16 + 8 * h + (e - 8)));
        efrag[mt][kc][e] = (_Float16)__expf(trans[i * NTAGS + mt * 16 + l16]);
      }

  // ---- init: score0 = start_transitions + emissions[0]; normalize.
  float z[4][8];
  float o;
#pragma unroll
  for (int mt = 0; mt < 4; ++mt)
#pragma unroll
    for (int q = 0; q < 2; ++q) {
      const int j0 = mt * 16 + 8 * h + 4 * q;
      v4f st = *(const v4f*)(startT + j0);
      v4f e0 = *(const v4f*)(em + (size_t)b * NTAGS + j0);
#pragma unroll
      for (int k = 0; k < 4; ++k) z[mt][4 * q + k] = (st[k] + e0[k]) * LOG2E;
    }
  {
    float mu = z[0][0];
#pragma unroll
    for (int mt = 0; mt < 4; ++mt)
#pragma unroll
      for (int v = 0; v < 8; ++v) mu = fmaxf(mu, z[mt][v]);
    mu = fmaxf(mu, __shfl_xor(mu, 16, 32));
#pragma unroll
    for (int mt = 0; mt < 4; ++mt)
#pragma unroll
      for (int v = 0; v < 8; ++v) z[mt][v] -= mu;
    o = mu;
  }

  // ---- emissions double buffer: preload t = 1
  v4f emv[2][4][2];
  int mkb[2];
  {
    const float* e1 = em + ((size_t)1 * BATCH + b) * NTAGS;
#pragma unroll
    for (int mt = 0; mt < 4; ++mt)
#pragma unroll
      for (int q = 0; q < 2; ++q)
        emv[1][mt][q] = *(const v4f*)(e1 + mt * 16 + 8 * h + 4 * q);
    mkb[1] = mask[1 * BATCH + b];
  }

#pragma unroll 2
  for (int t = 1; t < SEQ; ++t) {
    const int cur = t & 1, nx = (t + 1) & 1;

    // Prefetch t+4; load t+1 (clamped -> branchless) into the other buffer.
    const int tp = (t + 4 < SEQ) ? (t + 4) : (SEQ - 1);
    const int tn = (t + 1 < SEQ) ? (t + 1) : (SEQ - 1);
    __builtin_prefetch(em + ((size_t)tp * BATCH + b) * NTAGS + h * 32, 0, 3);
    {
      const float* en = em + ((size_t)tn * BATCH + b) * NTAGS;
#pragma unroll
      for (int mt = 0; mt < 4; ++mt)
#pragma unroll
        for (int q = 0; q < 2; ++q)
          emv[nx][mt][q] = *(const v4f*)(en + mt * 16 + 8 * h + 4 * q);
      mkb[nx] = mask[tn * BATCH + b];
    }

    // p = exp2(z) in (0,1]; pack to f16, scatter to LDS [b][tag].
    // No dependency on this step's loads -> starts immediately.
#pragma unroll
    for (int mt = 0; mt < 4; ++mt) {
      PackH8 pk;
#pragma unroll
      for (int q = 0; q < 4; ++q)
        pk.q[q] = __builtin_amdgcn_cvt_pkrtz(
            __builtin_amdgcn_exp2f(z[mt][2 * q]),
            __builtin_amdgcn_exp2f(z[mt][2 * q + 1]));
      *(h8*)&plds[w][l16][mt * 16 + 8 * h] = pk.v;
    }
    __builtin_amdgcn_wave_barrier();  // in-wave DS ordering covers the RAW

    // D[mt] = sum over 2 k-chunks of E^T(16x32) x P^T(32x16), f32 accumulate.
    v8f d[4];
#pragma unroll
    for (int mt = 0; mt < 4; ++mt)
#pragma unroll
      for (int v = 0; v < 8; ++v) d[mt][v] = 0.0f;

#pragma unroll
    for (int kc = 0; kc < 2; ++kc) {
      // B 32x16 layout: lane N=l16, elements 0..7 -> K = 32kc + 8h + e ; 8..15 -> K = 32kc+16+8h+(e-8).
      Frag16 bf;
      bf.part[0] = *(const h8*)&plds[w][l16][32 * kc + 8 * h];
      bf.part[1] = *(const h8*)&plds[w][l16][32 * kc + 16 + 8 * h];
#pragma unroll
      for (int mt = 0; mt < 4; ++mt)
        d[mt] = __builtin_amdgcn_wmma_f32_16x16x32_f16(
            false, efrag[mt][kc], false, bf.v, (short)0, d[mt], false, false);
    }
    __builtin_amdgcn_wave_barrier();  // keep next-iter stores after these loads

    // u_j = em_j*log2e + log2(q_j); mu = row max; z' = u - mu; o += mu.
    // asm pins keep the log/max chain unconditional (mask select stays
    // v_cndmask instead of a divergent saveexec region).
    float u[4][8];
#pragma unroll
    for (int mt = 0; mt < 4; ++mt)
#pragma unroll
      for (int v = 0; v < 8; ++v) {
        float uv = fmaf(emv[cur][mt][v >> 2][v & 3], LOG2E,
                        __builtin_amdgcn_logf(d[mt][v]));
        asm volatile("" : "+v"(uv));
        u[mt][v] = uv;
      }

    float mu = u[0][0];
#pragma unroll
    for (int mt = 0; mt < 4; ++mt)
#pragma unroll
      for (int v = 0; v < 8; ++v) mu = fmaxf(mu, u[mt][v]);
    mu = fmaxf(mu, __shfl_xor(mu, 16, 32));
    asm volatile("" : "+v"(mu));

    const int mk = mkb[cur];
#pragma unroll
    for (int mt = 0; mt < 4; ++mt)
#pragma unroll
      for (int v = 0; v < 8; ++v)
        z[mt][v] = mk ? (u[mt][v] - mu) : z[mt][v];
    o = mk ? (o + mu) : o;
  }

  // ---- denominator = ln2 * (o + log2sumexp2(z + endT*log2e))
#pragma unroll
  for (int mt = 0; mt < 4; ++mt)
#pragma unroll
    for (int q = 0; q < 2; ++q) {
      v4f et = *(const v4f*)(endT + mt * 16 + 8 * h + 4 * q);
#pragma unroll
      for (int k = 0; k < 4; ++k)
        z[mt][4 * q + k] = fmaf(et[k], LOG2E, z[mt][4 * q + k]);
    }

  float mx = z[0][0];
#pragma unroll
  for (int mt = 0; mt < 4; ++mt)
#pragma unroll
    for (int v = 0; v < 8; ++v) mx = fmaxf(mx, z[mt][v]);
  mx = fmaxf(mx, __shfl_xor(mx, 16, 32));

  float s = 0.0f;
#pragma unroll
  for (int mt = 0; mt < 4; ++mt)
#pragma unroll
    for (int v = 0; v < 8; ++v)
      s += __builtin_amdgcn_exp2f(z[mt][v] - mx);
  s += __shfl_xor(s, 16, 32);

  if (h == 0)
    den_out[b] = LN2F * (o + mx + __builtin_amdgcn_logf(s));
}

// ---------------------------------------------------------------------------
// Numerator: one thread per batch element; no loop-carried *memory* deps
// (mask is a contiguous prefix -> prev tag is just last iteration's register).
// Load sinking under the mask branch is desirable here (skips dead gathers).
// ---------------------------------------------------------------------------
__global__ __launch_bounds__(128)
void crf_score_kernel(const float* __restrict__ em,
                      const int*   __restrict__ tags,
                      const int*   __restrict__ mask,
                      const float* __restrict__ startT,
                      const float* __restrict__ endT,
                      const float* __restrict__ trans,
                      float*       __restrict__ num_out) {
  const int b = blockIdx.x * 128 + threadIdx.x;
  if (b >= BATCH) return;
  int prev = tags[b];
  int last = prev;
  float s = startT[prev] + em[(size_t)b * NTAGS + prev];
#pragma unroll 4
  for (int t = 1; t < SEQ; ++t) {
    const int   mk   = mask[t * BATCH + b];
    const int   tg   = tags[t * BATCH + b];
    const float term = trans[prev * NTAGS + tg] +
                       em[((size_t)t * BATCH + b) * NTAGS + tg];
    s += mk ? term : 0.0f;
    last = mk ? tg : last;
    prev = tg;  // valid whenever the term is actually used (prefix mask)
  }
  s += endT[last];
  num_out[b] = s;
}

// ---------------------------------------------------------------------------
// Deterministic final reduction: llh = sum_b (num[b] - den[b])
// ---------------------------------------------------------------------------
__global__ __launch_bounds__(32)
void crf_sum_kernel(const float* __restrict__ num,
                    const float* __restrict__ den,
                    float*       __restrict__ out) {
  const int lane = threadIdx.x & 31;
  float s = 0.0f;
  for (int i = lane; i < BATCH; i += 32) s += num[i] - den[i];
#pragma unroll
  for (int off = 16; off >= 1; off >>= 1) s += __shfl_xor(s, off, 32);
  if (lane == 0) out[0] = s;
}

extern "C" void kernel_launch(void* const* d_in, const int* in_sizes, int n_in,
                              void* d_out, int out_size, void* d_ws, size_t ws_size,
                              hipStream_t stream) {
  const float* em   = (const float*)d_in[0];   // (1024, 512, 64) f32
  const int*   tags = (const int*)  d_in[1];   // (1024, 512) int32
  const int*   mask = (const int*)  d_in[2];   // (1024, 512) int32
  const float* st   = (const float*)d_in[3];   // (64,)
  const float* et   = (const float*)d_in[4];   // (64,)
  const float* tr   = (const float*)d_in[5];   // (64, 64)
  float* out = (float*)d_out;
  float* den = (float*)d_ws;                   // [512]
  float* num = den + BATCH;                    // [512]

  crf_forward_kernel<<<4, 256, 0, stream>>>(em, mask, st, et, tr, den);
  crf_score_kernel<<<4, 128, 0, stream>>>(em, tags, mask, st, et, tr, num);
  crf_sum_kernel<<<1, 32, 0, stream>>>(num, den, out);
}
